// PLM_Graph_79834852098436
// MI455X (gfx1250) — matrix operand order repaired and checked by reading, not compiled, not run on hardware
//
#include <hip/hip_runtime.h>

// ---------------------------------------------------------------------------
// logits[B,L] = bert_output[B,H] @ W[L,H]^T + (dot(W[l],label_embed[l]) + b[l])
// B=16384, L=4096, H=1024, fp32 in/out.
//
// MI455X plan:
//  * fp32 -> (bf16 hi, bf16 lo) split PRE-COMPUTED once into d_ws (~80MB):
//    removes ~100 VALU/thread/iter from the GEMM hot loop and halves the
//    repeated-read footprint (bf16 A=32MB, W=8MB per copy; L2 is 192MB).
//  * GEMM: 3x v_wmma_f32_16x16x32_bf16 per tile (hi*hi + hi*lo + lo*hi, f32
//    accumulate ~= fp32 accuracy; dropped lo*lo term ~2^-18 relative).
//  * 128x128 block tile, BK=32, 8 waves (4x2), wave tile 32x64 = 2x4 WMMA tiles.
//  * LDS double buffer (2 x 32KB = 64KB), staged with
//    GLOBAL_LOAD_ASYNC_TO_LDS_B128 (ASYNCcnt-tracked, bypasses VGPRs);
//    register-staged copies as a fallback if the builtin is missing.
//    XOR chunk swizzle keeps fragment ds_load_b128 reads bank-conflict-free.
//  * Fallback path (small ws): round-1 fused-conversion kernel.
// ---------------------------------------------------------------------------

typedef __attribute__((ext_vector_type(16))) __bf16 v16bf;
typedef __attribute__((ext_vector_type(8)))  __bf16 v8bf;
typedef __attribute__((ext_vector_type(4)))  __bf16 v4bf;
typedef __attribute__((ext_vector_type(8)))  float  v8f;

#define B_DIM 16384
#define L_DIM 4096
#define H_DIM 1024

#define BM 128
#define BN 128
#define BK 32
#define NK (H_DIM / BK)
#define TILE_ELEMS (BM * BK)   // one hi-or-lo tile: 4096 bf16 = 8KB

#if __has_builtin(__builtin_amdgcn_global_load_async_to_lds_b128) && \
    __has_builtin(__builtin_amdgcn_s_wait_asynccnt)
#define USE_ASYNC_LDS 1
// signature (from hipcc diagnostic): (v4i AS1*, v4i AS3*, imm offset, imm cpol)
typedef int v4i __attribute__((vector_size(16)));
typedef __attribute__((address_space(1))) v4i as1_v4i;
typedef __attribute__((address_space(3))) v4i as3_v4i;
#else
#define USE_ASYNC_LDS 0
#endif

// ---- fp32 -> bf16 hi/lo split pre-pass -------------------------------------
__global__ __launch_bounds__(256) void split_bf16_kernel(
    const float* __restrict__ src, __bf16* __restrict__ hi,
    __bf16* __restrict__ lo, long long n4)
{
    const long long i = (long long)blockIdx.x * 256 + threadIdx.x;
    if (i >= n4) return;
    const float4 v = ((const float4*)src)[i];
    const float f[4] = {v.x, v.y, v.z, v.w};
    v4bf h, l;
    #pragma unroll
    for (int c = 0; c < 4; ++c) {
        h[c] = (__bf16)f[c];
        l[c] = (__bf16)(f[c] - (float)h[c]);
    }
    ((v4bf*)hi)[i] = h;
    ((v4bf*)lo)[i] = l;
}

// ---- per-label bias: bias[l] = dot(W[l], label_embed[l]) + b[l] -------------
__global__ __launch_bounds__(256) void label_bias_kernel(
    const float* __restrict__ W, const float* __restrict__ lemb,
    const float* __restrict__ b, float* __restrict__ bias_out)
{
    const int wave = threadIdx.x >> 5;
    const int lane = threadIdx.x & 31;
    const int l = blockIdx.x * 8 + wave;
    const float* wrow = W    + (size_t)l * H_DIM;
    const float* erow = lemb + (size_t)l * H_DIM;
    float s = 0.0f;
    #pragma unroll 4
    for (int h = lane; h < H_DIM; h += 32)
        s += wrow[h] * erow[h];
    #pragma unroll
    for (int off = 16; off > 0; off >>= 1)
        s += __shfl_down(s, off, 32);
    if (lane == 0) bias_out[l] = s + b[l];
}

// ---- main GEMM: pre-split bf16 operands, double-buffered LDS ----------------
__global__ __launch_bounds__(256) void plm_gemm_pre_kernel(
    const __bf16* __restrict__ Ahi, const __bf16* __restrict__ Alo,
    const __bf16* __restrict__ Whi, const __bf16* __restrict__ Wlo,
    const float* __restrict__ bias, float* __restrict__ C)
{
    // 2 buffers x 4 tiles (Ah,Al,Wh,Wl) x 4096 bf16 = 65536 B
    __shared__ __align__(16) __bf16 smem[2][4 * TILE_ELEMS];

    const int t      = threadIdx.x;
    const int blockN = blockIdx.x * BN;
    const int blockM = blockIdx.y * BM;

    const int wave = t >> 5;
    const int lane = t & 31;
    const int lo16 = lane & 15;
    const int sel  = lane >> 4;

    const int waveM = (wave >> 1) * 32;   // 4 waves down M
    const int waveN = (wave & 1) * 64;    // 2 waves across N

    v8f acc[2][4];
    #pragma unroll
    for (int i = 0; i < 2; ++i)
        #pragma unroll
        for (int j = 0; j < 4; ++j)
            acc[i][j] = (v8f)0.0f;

    // --- staging descriptors: each thread moves two 16B chunks per tile -----
    // chunk id c (0..511): row = c>>2, 16B sub-chunk cc = c&3
    // LDS chunk position is XOR-swizzled: cc' = cc ^ ((row>>2)&3)
    int gOff[2], lOff[2];
    #pragma unroll
    for (int j = 0; j < 2; ++j) {
        const int c   = t + j * 256;
        const int row = c >> 2;
        const int cc  = c & 3;
        gOff[j] = row * H_DIM + cc * 8;                       // + k0 later
        lOff[j] = row * BK + ((cc ^ ((row >> 2) & 3)) * 8);
    }
    const __bf16* gbase[4] = {
        Ahi + (size_t)blockM * H_DIM, Alo + (size_t)blockM * H_DIM,
        Whi + (size_t)blockN * H_DIM, Wlo + (size_t)blockN * H_DIM };

#if !USE_ASYNC_LDS
    v8bf stg[8];
#endif

    // ---- prologue: fill buffer 0 with tile k=0 -----------------------------
    #pragma unroll
    for (int arr = 0; arr < 4; ++arr)
        #pragma unroll
        for (int j = 0; j < 2; ++j) {
            const __bf16* g = gbase[arr] + gOff[j];
            __bf16* lp = &smem[0][arr * TILE_ELEMS + lOff[j]];
#if USE_ASYNC_LDS
            __builtin_amdgcn_global_load_async_to_lds_b128(
                (as1_v4i*)g, (as3_v4i*)lp, 0, 0);
#else
            *(v8bf*)lp = *(const v8bf*)g;
#endif
        }
#if USE_ASYNC_LDS
    __builtin_amdgcn_s_wait_asynccnt(0);
#endif
    __syncthreads();

    for (int kb = 0; kb < NK; ++kb) {
        const int  cur  = kb & 1;
        const bool more = (kb + 1) < NK;
        const int  k1   = (kb + 1) * BK;

        // ---- issue next tile into the other buffer -------------------------
        if (more) {
            #pragma unroll
            for (int arr = 0; arr < 4; ++arr)
                #pragma unroll
                for (int j = 0; j < 2; ++j) {
                    const __bf16* g = gbase[arr] + gOff[j] + k1;
#if USE_ASYNC_LDS
                    __bf16* lp = &smem[cur ^ 1][arr * TILE_ELEMS + lOff[j]];
                    __builtin_amdgcn_global_load_async_to_lds_b128(
                        (as1_v4i*)g, (as3_v4i*)lp, 0, 0);
#else
                    stg[arr * 2 + j] = *(const v8bf*)g;
#endif
                }
        }

        // ---- fragments from current buffer ---------------------------------
        const __bf16* sb = smem[cur];
        v16bf aHi[2], aLo[2], bHi[4], bLo[4];
        #pragma unroll
        for (int tm = 0; tm < 2; ++tm) {
            const int row = waveM + tm * 16 + lo16;
            const int g   = (row >> 2) & 3;
            const int c0  = ((sel) ^ g) * 8;      // K sel*8 .. +7
            const int c1  = ((2 + sel) ^ g) * 8;  // K 16+sel*8 .. +7
            const __bf16* ph = sb + 0 * TILE_ELEMS + row * BK;
            const __bf16* pl = sb + 1 * TILE_ELEMS + row * BK;
            v8bf h0 = *(const v8bf*)(ph + c0);
            v8bf h1 = *(const v8bf*)(ph + c1);
            v8bf l0 = *(const v8bf*)(pl + c0);
            v8bf l1 = *(const v8bf*)(pl + c1);
            aHi[tm] = __builtin_shufflevector(h0, h1, 0,1,2,3,4,5,6,7,8,9,10,11,12,13,14,15);
            aLo[tm] = __builtin_shufflevector(l0, l1, 0,1,2,3,4,5,6,7,8,9,10,11,12,13,14,15);
        }
        #pragma unroll
        for (int tn = 0; tn < 4; ++tn) {
            const int col = waveN + tn * 16 + lo16;
            const int g   = (col >> 2) & 3;
            const int c0  = ((sel) ^ g) * 8;      // K sel*8 .. +7
            const int c1  = ((2 + sel) ^ g) * 8;  // K 16+sel*8 .. +7
            const __bf16* ph = sb + 2 * TILE_ELEMS + col * BK;
            const __bf16* pl = sb + 3 * TILE_ELEMS + col * BK;
            v8bf h0 = *(const v8bf*)(ph + c0);
            v8bf h1 = *(const v8bf*)(ph + c1);
            v8bf l0 = *(const v8bf*)(pl + c0);
            v8bf l1 = *(const v8bf*)(pl + c1);
            bHi[tn] = __builtin_shufflevector(h0, h1, 0,1,2,3,4,5,6,7,8,9,10,11,12,13,14,15);
            bLo[tn] = __builtin_shufflevector(l0, l1, 0,1,2,3,4,5,6,7,8,9,10,11,12,13,14,15);
        }

        // ---- 24 WMMAs ------------------------------------------------------
        #pragma unroll
        for (int tm = 0; tm < 2; ++tm) {
            #pragma unroll
            for (int tn = 0; tn < 4; ++tn) {
                acc[tm][tn] = __builtin_amdgcn_wmma_f32_16x16x32_bf16(
                    false, aHi[tm], false, bHi[tn], (short)0, acc[tm][tn], false, false);
                acc[tm][tn] = __builtin_amdgcn_wmma_f32_16x16x32_bf16(
                    false, aHi[tm], false, bLo[tn], (short)0, acc[tm][tn], false, false);
                acc[tm][tn] = __builtin_amdgcn_wmma_f32_16x16x32_bf16(
                    false, aLo[tm], false, bHi[tn], (short)0, acc[tm][tn], false, false);
            }
        }

        if (more) {
#if !USE_ASYNC_LDS
            #pragma unroll
            for (int arr = 0; arr < 4; ++arr)
                #pragma unroll
                for (int j = 0; j < 2; ++j)
                    *(v8bf*)&smem[cur ^ 1][arr * TILE_ELEMS + lOff[j]] = stg[arr * 2 + j];
#else
            __builtin_amdgcn_s_wait_asynccnt(0);
#endif
            __syncthreads();
        }
    }

    // ---- epilogue: VGPR r -> M=r (+8 for lane half), lane -> N --------------
    #pragma unroll
    for (int tn = 0; tn < 4; ++tn) {
        const int col = blockN + waveN + tn * 16 + lo16;
        const float bv = bias[col];
        #pragma unroll
        for (int tm = 0; tm < 2; ++tm) {
            const int rbase = blockM + waveM + tm * 16 + sel * 8;
            #pragma unroll
            for (int r = 0; r < 8; ++r)
                C[(size_t)(rbase + r) * L_DIM + col] = acc[tm][tn][r] + bv;
        }
    }
}

// ---- fallback GEMM (round-1 style, converts in-loop; needs only bias ws) ----
#define LDK 40
__global__ __launch_bounds__(256) void plm_gemm_fused_kernel(
    const float* __restrict__ A, const float* __restrict__ W,
    const float* __restrict__ bias, float* __restrict__ C)
{
    __shared__ __align__(16) __bf16 Ah[BM * LDK];
    __shared__ __align__(16) __bf16 Al[BM * LDK];
    __shared__ __align__(16) __bf16 Wh[BN * LDK];
    __shared__ __align__(16) __bf16 Wl[BN * LDK];

    const int t      = threadIdx.x;
    const int blockN = blockIdx.x * BN;
    const int blockM = blockIdx.y * BM;
    const int wave = t >> 5, lane = t & 31, lo16 = lane & 15, sel = lane >> 4;
    const int waveM = (wave >> 1) * 32, waveN = (wave & 1) * 64;

    v8f acc[2][4];
    #pragma unroll
    for (int i = 0; i < 2; ++i)
        #pragma unroll
        for (int j = 0; j < 4; ++j) acc[i][j] = (v8f)0.0f;

    const int tk = (t & 7) * 4;
    const int tr = t >> 3;

    for (int k0 = 0; k0 < H_DIM; k0 += BK) {
        __syncthreads();
        #pragma unroll
        for (int r = 0; r < 4; ++r) {
            const int row = tr + r * 32;
            const float4 va = *(const float4*)(A + (size_t)(blockM + row) * H_DIM + k0 + tk);
            const float4 vw = *(const float4*)(W + (size_t)(blockN + row) * H_DIM + k0 + tk);
            const float af[4] = {va.x, va.y, va.z, va.w};
            const float wf[4] = {vw.x, vw.y, vw.z, vw.w};
            #pragma unroll
            for (int c = 0; c < 4; ++c) {
                const __bf16 ah = (__bf16)af[c];
                const __bf16 wh = (__bf16)wf[c];
                Ah[row * LDK + tk + c] = ah;
                Al[row * LDK + tk + c] = (__bf16)(af[c] - (float)ah);
                Wh[row * LDK + tk + c] = wh;
                Wl[row * LDK + tk + c] = (__bf16)(wf[c] - (float)wh);
            }
        }
        __syncthreads();

        v16bf aHi[2], aLo[2], bHi[4], bLo[4];
        #pragma unroll
        for (int tm = 0; tm < 2; ++tm) {
            const int row = waveM + tm * 16 + lo16;
            const __bf16* ph = &Ah[row * LDK + sel * 8];
            const __bf16* pl = &Al[row * LDK + sel * 8];
            v8bf h0 = *(const v8bf*)(ph), h1 = *(const v8bf*)(ph + 16);
            v8bf l0 = *(const v8bf*)(pl), l1 = *(const v8bf*)(pl + 16);
            aHi[tm] = __builtin_shufflevector(h0, h1, 0,1,2,3,4,5,6,7,8,9,10,11,12,13,14,15);
            aLo[tm] = __builtin_shufflevector(l0, l1, 0,1,2,3,4,5,6,7,8,9,10,11,12,13,14,15);
        }
        #pragma unroll
        for (int tn = 0; tn < 4; ++tn) {
            const int col = waveN + tn * 16 + lo16;
            const __bf16* ph = &Wh[col * LDK + sel * 8];
            const __bf16* pl = &Wl[col * LDK + sel * 8];
            v8bf h0 = *(const v8bf*)(ph), h1 = *(const v8bf*)(ph + 16);
            v8bf l0 = *(const v8bf*)(pl), l1 = *(const v8bf*)(pl + 16);
            bHi[tn] = __builtin_shufflevector(h0, h1, 0,1,2,3,4,5,6,7,8,9,10,11,12,13,14,15);
            bLo[tn] = __builtin_shufflevector(l0, l1, 0,1,2,3,4,5,6,7,8,9,10,11,12,13,14,15);
        }
        #pragma unroll
        for (int tm = 0; tm < 2; ++tm)
            #pragma unroll
            for (int tn = 0; tn < 4; ++tn) {
                acc[tm][tn] = __builtin_amdgcn_wmma_f32_16x16x32_bf16(
                    false, aHi[tm], false, bHi[tn], (short)0, acc[tm][tn], false, false);
                acc[tm][tn] = __builtin_amdgcn_wmma_f32_16x16x32_bf16(
                    false, aHi[tm], false, bLo[tn], (short)0, acc[tm][tn], false, false);
                acc[tm][tn] = __builtin_amdgcn_wmma_f32_16x16x32_bf16(
                    false, aLo[tm], false, bHi[tn], (short)0, acc[tm][tn], false, false);
            }
    }

    #pragma unroll
    for (int tn = 0; tn < 4; ++tn) {
        const int col = blockN + waveN + tn * 16 + lo16;
        const float bv = bias[col];
        #pragma unroll
        for (int tm = 0; tm < 2; ++tm) {
            const int rbase = blockM + waveM + tm * 16 + sel * 8;
            #pragma unroll
            for (int r = 0; r < 8; ++r)
                C[(size_t)(rbase + r) * L_DIM + col] = acc[tm][tn][r] + bv;
        }
    }
}

// ---------------------------------------------------------------------------
extern "C" void kernel_launch(void* const* d_in, const int* in_sizes, int n_in,
                              void* d_out, int out_size, void* d_ws, size_t ws_size,
                              hipStream_t stream) {
    (void)in_sizes; (void)n_in; (void)out_size;
    const float* bert = (const float*)d_in[0];  // [B,H]
    const float* lemb = (const float*)d_in[1];  // [L,H]
    const float* W    = (const float*)d_in[2];  // [L,H]
    const float* bvec = (const float*)d_in[3];  // [L]
    float* out = (float*)d_out;                 // [B,L]

    const size_t nA = (size_t)B_DIM * H_DIM;    // 16M elems
    const size_t nW = (size_t)L_DIM * H_DIM;    // 4M elems
    const size_t need = (2 * nA + 2 * nW) * sizeof(__bf16) + L_DIM * sizeof(float);
    const dim3 grid(L_DIM / BN, B_DIM / BM);    // (32, 128)

    if (ws_size >= need) {
        __bf16* Ahi = (__bf16*)d_ws;
        __bf16* Alo = Ahi + nA;
        __bf16* Whi = Alo + nA;
        __bf16* Wlo = Whi + nW;
        float*  bias = (float*)(Wlo + nW);

        split_bf16_kernel<<<(int)(nA / 4 / 256), 256, 0, stream>>>(bert, Ahi, Alo, (long long)(nA / 4));
        split_bf16_kernel<<<(int)(nW / 4 / 256), 256, 0, stream>>>(W,    Whi, Wlo, (long long)(nW / 4));
        label_bias_kernel<<<L_DIM / 8, 256, 0, stream>>>(W, lemb, bvec, bias);
        plm_gemm_pre_kernel<<<grid, 256, 0, stream>>>(Ahi, Alo, Whi, Wlo, bias, out);
    } else {
        float* bias = (float*)d_ws;             // 16 KB
        label_bias_kernel<<<L_DIM / 8, 256, 0, stream>>>(W, lemb, bvec, bias);
        plm_gemm_fused_kernel<<<grid, 256, 0, stream>>>(bert, W, bias, out);
    }
}
